// Loss_41085657154198
// MI455X (gfx1250) — matrix operand order repaired and checked by reading, not compiled, not run on hardware
//
#include <hip/hip_runtime.h>
#include <hip/hip_bf16.h>

// ---------------------------------------------------------------------------
// Fused InfoNCE + mutual-NN matching for 12288x12288x128 similarity.
// Two WMMA passes (row stats of S, row stats of S^T), S never materialized.
// Double-buffered B fragments, SADDR+imm addressing, no WAR nop bubbles.
// ---------------------------------------------------------------------------

#define DIMK 128
constexpr int NSPLIT = 8;      // column-range splits per pass (occupancy)
constexpr float SCALE = 10.0f; // 1/T with T = 0.1 (both sides get 1/sqrt(T))
constexpr float EXP2SCALE = 14.426950408889634f; // 10 * log2(e)

typedef __attribute__((ext_vector_type(16))) _Float16 v16h;
typedef __attribute__((ext_vector_type(8)))  _Float16 v8h;
typedef __attribute__((ext_vector_type(4)))  _Float16 v4h;
typedef __attribute__((ext_vector_type(8)))  float    v8f;

// ----------------------------- fp32 -> f16 cast ----------------------------
__global__ __launch_bounds__(256) void cvt_f32_f16(const float* __restrict__ in,
                                                   _Float16* __restrict__ out, int n) {
    int i = blockIdx.x * blockDim.x + threadIdx.x;
    if (i < n) out[i] = (_Float16)in[i];
}

// ------------------- fused GEMM + row logsumexp/argmax ---------------------
// A: [NA][128] row-major f16 (rows of the output), B: [NB][128] f16 (cols).
// Each wave owns 16 rows and one 1/NSPLIT slice of the NB columns.
// Writes per-split partials: psum (sum of exp(10*S)), pval/pidx (argmax, raw).
__global__ __launch_bounds__(256)
void gemm_stats_kernel(const _Float16* __restrict__ A, const _Float16* __restrict__ B,
                       int NA, int NB,
                       float* __restrict__ psum, float* __restrict__ pval,
                       int* __restrict__ pidx) {
    const int lane = threadIdx.x & 31;
    const int wave = threadIdx.x >> 5;
    const int rowblk = blockIdx.x * 8 + wave;
    const int i0 = rowblk * 16;
    if (i0 >= NA) return;                       // wave-uniform: EXEC stays all-1s

    const int split = blockIdx.y;
    const int jlen  = NB / NSPLIT;
    const int jbeg  = split * jlen;
    const int jend  = jbeg + jlen;

    const int lrow = lane & 15;
    const int lhi  = lane >> 4;                 // 0: low half-wave, 1: high

    // Preload all 4 A fragments (K = 0..127), reused across the whole j loop.
    // A 16x32 k order used here (contiguous, same as B): lanes 0-15: K in {0..15} of kk*32,
    // lanes 16-31: K in {16..31}.
    v16h afrag[4];
    {
        const _Float16* arow = A + (size_t)(i0 + lrow) * DIMK;
        #pragma unroll
        for (int kk = 0; kk < 4; ++kk) {
            const int kbase = kk * 32 + lhi * 16;
            v8h lo = *(const v8h*)(arow + kbase);
            v8h hi = *(const v8h*)(arow + kbase + 8);
            v16h a;
            #pragma unroll
            for (int e = 0; e < 8; ++e) { a[e] = lo[e]; a[e + 8] = hi[e]; }
            afrag[kk] = a;
        }
    }

    // Per-lane partial stats: accumulator VGPR r holds row (i0 + r + 8*lhi),
    // column class (lane&15). S <= 10 so sum(exp(10*S)) fits fp32: no max shift.
    float rsum[8], rbest[8];
    int   ribest[8];
    #pragma unroll
    for (int r = 0; r < 8; ++r) { rsum[r] = 0.f; rbest[r] = -3.0e38f; ribest[r] = 0; }

    // All B addresses = uniform base + per-lane u32 offset + small immediates:
    //   byte = row*256 + kk*64 + lhi*32, row = j0 + t*16 + lrow
    // -> voffset advances by a constant 16384 per 64-column chunk; tile/kk
    //    deltas (t*4096 + kk*64) are instruction immediates.
    const char* Bc = (const char*)B;
    uint32_t boff = ((uint32_t)(jbeg + lrow) << 8) + (uint32_t)(lhi << 5);

    v16h bbuf[2][4];
    auto loadB = [&](uint32_t base, int kk, v16h* dst) {
        #pragma unroll
        for (int t = 0; t < 4; ++t)
            dst[t] = *(const v16h*)(Bc + base + (uint32_t)(t * 4096 + kk * 64));
    };

    loadB(boff, 0, bbuf[0]);                   // prime the pipeline
    for (int j0 = jbeg; j0 < jend; j0 += 64) {
        v8f z = {0.f, 0.f, 0.f, 0.f, 0.f, 0.f, 0.f, 0.f};
        v8f c[4] = {z, z, z, z};
        #pragma unroll
        for (int kk = 0; kk < 4; ++kk) {
            // Issue next K-step's (or next chunk's) B loads before consuming
            // this step's fragments: loads overlap the WMMA pipe, and the
            // other buffer avoids WMMA->VALU WAR hazard NOPs.
            if (kk < 3) loadB(boff, kk + 1, bbuf[(kk + 1) & 1]);
            else        loadB(boff + 16384u, 0, bbuf[0]);   // harmless overfetch on last chunk (stays in ws)
            #pragma unroll
            for (int t = 0; t < 4; ++t) {
                c[t] = __builtin_amdgcn_wmma_f32_16x16x32_f16(
                    false, afrag[kk], false, bbuf[kk & 1][t], (short)0, c[t], false, false);
            }
        }
        // Online stats (VALU/trans co-executes with the WMMA pipe).
        // argmax compares raw accumulators (monotonic under the +10 scale;
        // only the index is consumed downstream).
        #pragma unroll
        for (int t = 0; t < 4; ++t) {
            const int j = j0 + t * 16 + lrow;
            #pragma unroll
            for (int r = 0; r < 8; ++r) {
                float cv = c[t][r];
                rsum[r] += __builtin_amdgcn_exp2f(cv * EXP2SCALE);
                ribest[r] = (cv > rbest[r]) ? j : ribest[r];
                rbest[r]  = fmaxf(rbest[r], cv);
            }
        }
        boff += 16384u;
    }

    // Reduce across the 16 lanes of each half-wave (xor offsets stay in-group).
    #pragma unroll
    for (int r = 0; r < 8; ++r) {
        float s = rsum[r]; float bv = rbest[r]; int bi = ribest[r];
        #pragma unroll
        for (int off = 8; off >= 1; off >>= 1) {
            s += __shfl_xor(s, off, 32);
            float ov = __shfl_xor(bv, off, 32);
            int   oi = __shfl_xor(bi, off, 32);
            bool take = (ov > bv) || (ov == bv && oi < bi);
            bv = take ? ov : bv;
            bi = take ? oi : bi;
        }
        rsum[r] = s; rbest[r] = bv; ribest[r] = bi;
    }
    if ((lane & 15) == 0) {
        const int rowbase = i0 + lhi * 8;
        #pragma unroll
        for (int r = 0; r < 8; ++r) {
            size_t o = (size_t)split * NA + rowbase + r;
            psum[o] = rsum[r];
            pval[o] = rbest[r];
            pidx[o] = ribest[r];
        }
    }
}

// ----------------- merge per-split partials -> lse / argmax ----------------
__global__ __launch_bounds__(256)
void combine_kernel(const float* __restrict__ psum, const float* __restrict__ pval,
                    const int* __restrict__ pidx, int NA,
                    float* __restrict__ lse, int* __restrict__ best) {
    int i = blockIdx.x * blockDim.x + threadIdx.x;
    if (i >= NA) return;
    float s = 0.f, bv = -3.0e38f; int bi = 0;
    #pragma unroll
    for (int p = 0; p < NSPLIT; ++p) {
        size_t o = (size_t)p * NA + i;
        s += psum[o];
        float v = pval[o]; int ix = pidx[o];
        if (v > bv || (v == bv && ix < bi)) { bv = v; bi = ix; }
    }
    lse[i]  = __logf(s);   // natural-log LSE: values <= 10, no max shift needed
    best[i] = bi;
}

// ---------------- pos[i] = scale * <A[i], B[corr[i]]> (or 0) ---------------
__global__ __launch_bounds__(256)
void pos_kernel(const _Float16* __restrict__ A, const _Float16* __restrict__ B,
                const int* __restrict__ corr, float* __restrict__ pos, int n) {
    int lane = threadIdx.x & 31;
    int row  = (blockIdx.x * blockDim.x + threadIdx.x) >> 5;
    if (row >= n) return;
    int c = corr[row];
    float acc = 0.f;
    if (c >= 0) {
        v4h a = *(const v4h*)(A + (size_t)row * DIMK + lane * 4);
        v4h b = *(const v4h*)(B + (size_t)c   * DIMK + lane * 4);
        #pragma unroll
        for (int e = 0; e < 4; ++e) acc += (float)a[e] * (float)b[e];
    }
    #pragma unroll
    for (int off = 16; off >= 1; off >>= 1) acc += __shfl_xor(acc, off, 32);
    if (lane == 0) pos[row] = (c >= 0) ? acc * SCALE : 0.0f;
}

// -------- final scalar reductions: losses, precision, recall (1 block) -----
__global__ __launch_bounds__(256)
void final_kernel(const float* __restrict__ lse0, const float* __restrict__ pos0,
                  const int* __restrict__ corr0, const int* __restrict__ best0,
                  const float* __restrict__ lse1, const float* __restrict__ pos1,
                  const int* __restrict__ corr1, const int* __restrict__ best1,
                  const float* __restrict__ lg0,  const float* __restrict__ lg1,
                  int n, int m, float* __restrict__ out) {
    __shared__ float rl0[256], rl1[256];
    __shared__ int   rc0[256], rc1[256], rtp[256], rpr[256];
    const int tid = threadIdx.x;
    float l0 = 0.f, l1 = 0.f; int c0 = 0, c1 = 0, tp = 0, pr = 0;

    for (int i = tid; i < n; i += 256) {
        bool m0 = corr0[i] >= 0;
        if (m0) { l0 += lse0[i] - pos0[i]; ++c0; }
        int b0 = best0[i];
        bool mutual = (best1[b0] == i);
        bool predicted = mutual && (lg0[i] >= 0.f) && (lg1[b0] >= 0.f);
        bool correct = m0 && (b0 == corr0[i]);
        pr += predicted ? 1 : 0;
        tp += (predicted && correct) ? 1 : 0;
    }
    for (int j = tid; j < m; j += 256) {
        if (corr1[j] >= 0) { l1 += lse1[j] - pos1[j]; ++c1; }
    }
    rl0[tid] = l0; rl1[tid] = l1; rc0[tid] = c0; rc1[tid] = c1;
    rtp[tid] = tp; rpr[tid] = pr;
    __syncthreads();
    for (int s = 128; s > 0; s >>= 1) {
        if (tid < s) {
            rl0[tid] += rl0[tid + s]; rl1[tid] += rl1[tid + s];
            rc0[tid] += rc0[tid + s]; rc1[tid] += rc1[tid + s];
            rtp[tid] += rtp[tid + s]; rpr[tid] += rpr[tid + s];
        }
        __syncthreads();
    }
    if (tid == 0) {
        out[0] = rl0[0] / (float)max(rc0[0], 1);
        out[1] = rl1[0] / (float)max(rc1[0], 1);
        out[2] = (float)rtp[0] / (float)max(rpr[0], 1);
        out[3] = (float)rtp[0] / (float)max(rc0[0], 1);
    }
}

// ---------------------------------------------------------------------------
extern "C" void kernel_launch(void* const* d_in, const int* in_sizes, int n_in,
                              void* d_out, int out_size, void* d_ws, size_t ws_size,
                              hipStream_t stream) {
    const float* desc0 = (const float*)d_in[0];   // [N,128]
    const float* desc1 = (const float*)d_in[1];   // [M,128]
    const int*   corr0 = (const int*)d_in[2];     // [N]
    const int*   corr1 = (const int*)d_in[3];     // [M]
    const float* lg0   = (const float*)d_in[4];   // [N]
    const float* lg1   = (const float*)d_in[5];   // [M]
    float* out = (float*)d_out;                   // 4 scalars

    const int N = in_sizes[2];
    const int M = in_sizes[3];
    (void)n_in; (void)out_size; (void)ws_size;

    // Workspace carve-up (256B-aligned chunks; ~9 MB total).
    char* p = (char*)d_ws;
    auto take = [&](size_t bytes) { char* r = p; p += (bytes + 255) & ~(size_t)255; return r; };
    _Float16* d0h  = (_Float16*)take((size_t)N * DIMK * 2);
    _Float16* d1h  = (_Float16*)take((size_t)M * DIMK * 2);
    float* psum0 = (float*)take((size_t)NSPLIT * N * 4);
    float* pval0 = (float*)take((size_t)NSPLIT * N * 4);
    int*   pidx0 = (int*)  take((size_t)NSPLIT * N * 4);
    float* psum1 = (float*)take((size_t)NSPLIT * M * 4);
    float* pval1 = (float*)take((size_t)NSPLIT * M * 4);
    int*   pidx1 = (int*)  take((size_t)NSPLIT * M * 4);
    float* lse0  = (float*)take((size_t)N * 4);
    float* lse1  = (float*)take((size_t)M * 4);
    int*   best0 = (int*)  take((size_t)N * 4);
    int*   best1 = (int*)  take((size_t)M * 4);
    float* pos0  = (float*)take((size_t)N * 4);
    float* pos1  = (float*)take((size_t)M * 4);

    // 1) cast descriptors to f16 (unit-norm data: f16 mantissa is ample)
    cvt_f32_f16<<<(N * DIMK + 255) / 256, 256, 0, stream>>>(desc0, d0h, N * DIMK);
    cvt_f32_f16<<<(M * DIMK + 255) / 256, 256, 0, stream>>>(desc1, d1h, M * DIMK);

    // 2) pass A: row stats of S;  pass B: row stats of S^T (= column stats)
    gemm_stats_kernel<<<dim3(N / 128, NSPLIT), 256, 0, stream>>>(
        d0h, d1h, N, M, psum0, pval0, pidx0);
    gemm_stats_kernel<<<dim3(M / 128, NSPLIT), 256, 0, stream>>>(
        d1h, d0h, M, N, psum1, pval1, pidx1);

    // 3) merge split partials -> lse / argmax
    combine_kernel<<<(N + 255) / 256, 256, 0, stream>>>(psum0, pval0, pidx0, N, lse0, best0);
    combine_kernel<<<(M + 255) / 256, 256, 0, stream>>>(psum1, pval1, pidx1, M, lse1, best1);

    // 4) positive-pair scores
    pos_kernel<<<(N * 32 + 255) / 256, 256, 0, stream>>>(d0h, d1h, corr0, pos0, N);
    pos_kernel<<<(M * 32 + 255) / 256, 256, 0, stream>>>(d1h, d0h, corr1, pos1, M);

    // 5) losses + precision/recall
    final_kernel<<<1, 256, 0, stream>>>(lse0, pos0, corr0, best0,
                                        lse1, pos1, corr1, best1,
                                        lg0, lg1, N, M, out);
}
